// InfSAAttention_2216203125104
// MI455X (gfx1250) — matrix operand order repaired and checked by reading, not compiled, not run on hardware
//
#include <hip/hip_runtime.h>
#include <hip/hip_bf16.h>

typedef _Float16 half_t;
typedef __attribute__((ext_vector_type(16))) _Float16 v16h;
typedef __attribute__((ext_vector_type(8)))  _Float16 v8h;
typedef __attribute__((ext_vector_type(2)))  _Float16 v2h;
typedef __attribute__((ext_vector_type(8)))  float    v8f;

#define CB  2
#define CN  2048
#define CE  1024
#define CH  16
#define CD  64
#define CM  (CB*CN)   // 4096 rows

// ---------------------------------------------------------------------------
// helpers: build WMMA A / B fragments from k-major storage
//   A-frag (16x32): lane&15 = row; K chunks {0-7,16-23} / {8-15,24-31} by half
//   B-frag (32x16): lane&15 = column (a row of W [n,k]); K chunks {0-7,16-23} / {8-15,24-31} by half
// ---------------------------------------------------------------------------
__device__ __forceinline__ v16h load_a_frag(const half_t* __restrict__ base,
                                            int lda, int row0, int k0, int lane) {
    const int kasel = (lane & 16) ? 8 : 0;
    const half_t* p = base + (size_t)(row0 + (lane & 15)) * lda + k0;
    v8h lo = *(const v8h*)(p + kasel);
    v8h hi = *(const v8h*)(p + 16 + kasel);
    return __builtin_shufflevector(lo, hi, 0,1,2,3,4,5,6,7,8,9,10,11,12,13,14,15);
}

__device__ __forceinline__ v16h load_b_frag(const half_t* __restrict__ base,
                                            int ldb, int col0, int k0, int lane) {
    const int kbsel = (lane & 16) ? 8 : 0;
    const half_t* p = base + (size_t)(col0 + (lane & 15)) * ldb + k0;
    v8h lo = *(const v8h*)(p + kbsel);
    v8h hi = *(const v8h*)(p + 16 + kbsel);
    return __builtin_shufflevector(lo, hi, 0,1,2,3,4,5,6,7,8,9,10,11,12,13,14,15);
}

__device__ __forceinline__ v8f wmma_f16(v16h a, v16h b, v8f c) {
    return __builtin_amdgcn_wmma_f32_16x16x32_f16(false, a, false, b,
                                                  (short)0, c, false, false);
}

// ---------------------------------------------------------------------------
// fp32 -> fp16 conversion (x and the four weight matrices)
// ---------------------------------------------------------------------------
__global__ void cvt_f32_to_f16(const float* __restrict__ src,
                               half_t* __restrict__ dst, int n) {
    int i = blockIdx.x * blockDim.x + threadIdx.x;
    int stride = gridDim.x * blockDim.x;
    for (; i < n; i += stride) dst[i] = (half_t)src[i];
}

// ---------------------------------------------------------------------------
// GEMM: C[m,n] = (sum_k A16[m,k] * W16[n,k] + bias[n]) * alpha
// A: [M,K] k-major f16.  W: [Nout,K] k-major f16 (torch Linear weight).
// MODE 0: f16 out, head-major  [b,h,nseq,d]      (Q, K)
// MODE 1: f16 out, transposed  [b,h,d,nseq]      (V)
// MODE 2: f32 out, row-major   [m, Nout]         (final projection -> d_out)
// Block: 256 threads (8 waves), waves 2(M) x 4(N).
// Wave tile 32x64: 2 A-frags x 4 B-frags -> 8 WMMAs/k-step, B reused 2x.
// ---------------------------------------------------------------------------
template <int MODE>
__global__ void gemm_wmma(const half_t* __restrict__ A,
                          const half_t* __restrict__ W,
                          const float*  __restrict__ bias,
                          void* __restrict__ dst, float alpha) {
    const int lane = threadIdx.x & 31;
    const int wave = threadIdx.x >> 5;
    const int row0 = blockIdx.y * 64  + (wave & 1) * 32;
    const int col0 = blockIdx.x * 256 + (wave >> 1) * 64;

    const half_t* pa = A + (size_t)(row0 + (lane & 15)) * CE;
    const half_t* pb = W + (size_t)(col0 + (lane & 15)) * CE;

    v8f c[2][4] = {};
    for (int k0 = 0; k0 < CE; k0 += 32) {
        // prefetch next k-tile toward the WGP while WMMAs drain
        if (k0 + 64 < CE) {
            __builtin_prefetch(pa + k0 + 64, 0, 3);
            __builtin_prefetch(pb + k0 + 64, 0, 3);
        }
        v16h a0 = load_a_frag(A, CE, row0,      k0, lane);
        v16h a1 = load_a_frag(A, CE, row0 + 16, k0, lane);
#pragma unroll
        for (int t = 0; t < 4; ++t) {
            v16h b = load_b_frag(W, CE, col0 + t * 16, k0, lane);
            c[0][t] = wmma_f16(a0, b, c[0][t]);
            c[1][t] = wmma_f16(a1, b, c[1][t]);
        }
    }

    const int mhalf = (lane >> 4) << 3;
#pragma unroll
    for (int mt = 0; mt < 2; ++mt) {
#pragma unroll
        for (int t = 0; t < 4; ++t) {
            const int col = col0 + t * 16 + (lane & 15);
            const float bb = bias[col];
#pragma unroll
            for (int r = 0; r < 8; ++r) {
                const int m = row0 + mt * 16 + r + mhalf;
                const float v = (c[mt][t][r] + bb) * alpha;
                if (MODE == 2) {
                    ((float*)dst)[(size_t)m * CE + col] = v;
                } else {
                    const int bI = m >> 11, nseq = m & (CN - 1);
                    const int h  = col >> 6, d    = col & (CD - 1);
                    const half_t hv = (half_t)v;
                    if (MODE == 0)
                        ((half_t*)dst)[(((size_t)(bI * CH + h) * CN) + nseq) * CD + d] = hv;
                    else
                        ((half_t*)dst)[(((size_t)(bI * CH + h) * CD) + d) * CN + nseq] = hv;
                }
            }
        }
    }
}

// ---------------------------------------------------------------------------
// Attention: out[q,d] = sum_s (Q[q]·K[s]) * rho^|q-s| * V[s,d]
// (1/sqrt(D) already folded into Q.)
// Q,K: [b,h,s,d] f16.  Vt: [b,h,d,s] f16.  ctx out: [b,nseq,E] f16.
// Block: 128 threads (4 waves); each wave owns 32 query rows x D=64 output,
// so K and V fragments are each reused by two score/A tiles (16 WMMA/s-step).
// __launch_bounds__(128,1): one wave/SIMD, full VGPR file -> no scratch spill.
// Score repack: each lane packs (col j, col j+16) of one row into a v2h and
// does ONE ds_store_b32; reader loads its pair-run and de-interleaves in
// registers. Same-wave DS ops are in-order on CDNA5 -> no barrier needed.
// ---------------------------------------------------------------------------
__global__ __launch_bounds__(128, 1)
void attn_wmma(const half_t* __restrict__ q16,
               const half_t* __restrict__ k16,
               const half_t* __restrict__ vt16,
               const float*  __restrict__ rho_logit,
               half_t* __restrict__ ctx16) {
    // pair-interleaved score buffer: [wave][row][pair j] = (S[row,j], S[row,j+16])
    __shared__ __align__(32) v2h sbuf[4][32][16];

    const int lane = threadIdx.x & 31;
    const int wave = threadIdx.x >> 5;
    const int bh = blockIdx.y;
    const int b = bh >> 4, h = bh & 15;
    const int q0 = blockIdx.x * 128 + wave * 32;

    const half_t* qb = q16  + (size_t)bh * CN * CD;
    const half_t* kb = k16  + (size_t)bh * CN * CD;
    const half_t* vb = vt16 + (size_t)bh * CD * CN;

    const float lg  = rho_logit[0];
    const float rho = 1.0f / (1.0f + __builtin_amdgcn_exp2f(-lg * 1.4426950408889634f));
    const float l2r = __builtin_amdgcn_logf(rho);   // log2(rho)

    const int rloc  = lane & 15;
    const int mhalf = (lane >> 4) << 3;
    const int kasel = (lane & 16) ? 8 : 0;

    // Q fragments: 2 row-tiles x 2 k-steps (D = 64)
    v16h aq[2][2];
#pragma unroll
    for (int m2 = 0; m2 < 2; ++m2)
#pragma unroll
        for (int t = 0; t < 2; ++t)
            aq[m2][t] = load_a_frag(qb, CD, q0 + m2 * 16, t * 32, lane);

    const half_t* pkpre = kb + (size_t)rloc * CD;
    const half_t* pvpre = vb + (size_t)rloc * CN;

    v8f acc[2][4] = {};

    for (int s0 = 0; s0 < CN; s0 += 32) {
        if (s0 + 32 < CN) {
            __builtin_prefetch(pkpre + (size_t)(s0 + 32) * CD, 0, 3);
            __builtin_prefetch(pvpre + s0 + 32, 0, 3);
        }

        // ---- scores S[32q x 32s] = Q * K^T (K frags reused by both q tiles)
        v8f sc[2][2] = {};
#pragma unroll
        for (int st = 0; st < 2; ++st)
#pragma unroll
            for (int t = 0; t < 2; ++t) {
                v16h bk = load_b_frag(kb, CD, s0 + st * 16, t * 32, lane);
                sc[0][st] = wmma_f16(aq[0][t], bk, sc[0][st]);
                sc[1][st] = wmma_f16(aq[1][t], bk, sc[1][st]);
            }

        // ---- distance decay, f32 -> f16 pair, one b32 LDS store per row ---
#pragma unroll
        for (int m2 = 0; m2 < 2; ++m2)
#pragma unroll
            for (int r = 0; r < 8; ++r) {
                const int i  = q0 + m2 * 16 + r + mhalf;
                const int j0 = s0 + rloc;
                const float d0 = fabsf((float)(i - j0));
                const float d1 = fabsf((float)(i - j0 - 16));
                v2h hv;
                hv.x = (half_t)(sc[m2][0][r] * __builtin_amdgcn_exp2f(d0 * l2r));
                hv.y = (half_t)(sc[m2][1][r] * __builtin_amdgcn_exp2f(d1 * l2r));
                sbuf[wave][m2 * 16 + r + mhalf][rloc] = hv;
            }

        // ---- reload as WMMA A-fragments (same wave: DS is in-order) -------
        v16h a2[2];
#pragma unroll
        for (int m2 = 0; m2 < 2; ++m2) {
            const v2h* pp = &sbuf[wave][m2 * 16 + rloc][kasel];
            v16h raw = *(const v16h*)pp;   // pairs kasel..kasel+7
            // de-interleave: evens = K kasel..kasel+7, odds = K+16
            a2[m2] = __builtin_shufflevector(raw, raw,
                                             0,2,4,6,8,10,12,14,
                                             1,3,5,7,9,11,13,15);
        }

        // ---- context += S * V  (V frags reused by both q tiles) -----------
#pragma unroll
        for (int dt = 0; dt < 4; ++dt) {
            v16h bv = load_b_frag(vb, CN, dt * 16, s0, lane);
            acc[0][dt] = wmma_f16(a2[0], bv, acc[0][dt]);
            acc[1][dt] = wmma_f16(a2[1], bv, acc[1][dt]);
        }
    }

    // ---- store context as [b, nseq, E] f16 for the output projection -----
#pragma unroll
    for (int m2 = 0; m2 < 2; ++m2)
#pragma unroll
        for (int dt = 0; dt < 4; ++dt)
#pragma unroll
            for (int r = 0; r < 8; ++r) {
                const int m   = q0 + m2 * 16 + r + mhalf;
                const int col = h * CD + dt * 16 + rloc;
                ctx16[((size_t)(b * CN + m)) * CE + col] = (half_t)acc[m2][dt][r];
            }
}

// ---------------------------------------------------------------------------
extern "C" void kernel_launch(void* const* d_in, const int* in_sizes, int n_in,
                              void* d_out, int out_size, void* d_ws, size_t ws_size,
                              hipStream_t stream) {
    (void)in_sizes; (void)n_in; (void)out_size; (void)ws_size;

    const float* x   = (const float*)d_in[0];
    const float* Wq  = (const float*)d_in[1];
    const float* bq  = (const float*)d_in[2];
    const float* Wk  = (const float*)d_in[3];
    const float* bk  = (const float*)d_in[4];
    const float* Wv  = (const float*)d_in[5];
    const float* bv  = (const float*)d_in[6];
    const float* Wo  = (const float*)d_in[7];
    const float* bo  = (const float*)d_in[8];
    const float* rho = (const float*)d_in[9];

    const size_t MB = 1u << 20;
    char* ws = (char*)d_ws;
    half_t* x16   = (half_t*)(ws + 0 * MB);   // 8 MB
    half_t* wq16  = (half_t*)(ws + 8 * MB);   // 2 MB
    half_t* wk16  = (half_t*)(ws + 10 * MB);  // 2 MB
    half_t* wv16  = (half_t*)(ws + 12 * MB);  // 2 MB
    half_t* wo16  = (half_t*)(ws + 14 * MB);  // 2 MB
    half_t* q16   = (half_t*)(ws + 16 * MB);  // 8 MB  [b,h,n,d]
    half_t* k16   = (half_t*)(ws + 24 * MB);  // 8 MB  [b,h,n,d]
    half_t* vt16  = (half_t*)(ws + 32 * MB);  // 8 MB  [b,h,d,n]
    half_t* ctx16 = (half_t*)(ws + 40 * MB);  // 8 MB  [b,n,E]

    // 1) fp32 -> fp16 conversions
    cvt_f32_to_f16<<<1024, 256, 0, stream>>>(x,  x16,  CM * CE);
    cvt_f32_to_f16<<<512,  256, 0, stream>>>(Wq, wq16, CE * CE);
    cvt_f32_to_f16<<<512,  256, 0, stream>>>(Wk, wk16, CE * CE);
    cvt_f32_to_f16<<<512,  256, 0, stream>>>(Wv, wv16, CE * CE);
    cvt_f32_to_f16<<<512,  256, 0, stream>>>(Wo, wo16, CE * CE);

    // 2) QKV projections (1/sqrt(D) folded into Q)
    dim3 ggrid(CE / 256, CM / 64);
    gemm_wmma<0><<<ggrid, 256, 0, stream>>>(x16, wq16, bq, q16,  0.125f);
    gemm_wmma<0><<<ggrid, 256, 0, stream>>>(x16, wk16, bk, k16,  1.0f);
    gemm_wmma<1><<<ggrid, 256, 0, stream>>>(x16, wv16, bv, vt16, 1.0f);

    // 3) decayed attention
    dim3 agrid(CN / 128, CB * CH);
    attn_wmma<<<agrid, 128, 0, stream>>>(q16, k16, vt16, rho, ctx16);

    // 4) output projection -> fp32 d_out
    gemm_wmma<2><<<ggrid, 256, 0, stream>>>(ctx16, wo16, bo, d_out, 1.0f);
}
